// SelfAttnModel_36266703847932
// MI455X (gfx1250) — matrix operand order repaired and checked by reading, not compiled, not run on hardware
//
#include <hip/hip_runtime.h>
#include <cstdint>

typedef __attribute__((ext_vector_type(16))) __bf16 v16bf;
typedef __attribute__((ext_vector_type(8)))  float  v8f;
typedef __attribute__((ext_vector_type(4)))  unsigned int u32x4;
typedef __attribute__((ext_vector_type(8)))  int i32x8;
typedef __attribute__((ext_vector_type(4)))  int i32x4;

union V16BF {
  v16bf v;
  unsigned short u[16];
  uint32_t w[8];
};

__device__ __forceinline__ unsigned short f2bf(float f) {
  uint32_t x = __builtin_bit_cast(uint32_t, f);
  uint32_t r = (x + 0x7FFFu + ((x >> 16) & 1u)) >> 16;  // round-to-nearest-even
  return (unsigned short)r;
}

#if __has_builtin(__builtin_amdgcn_cvt_pk_bf16_f32)
typedef __attribute__((ext_vector_type(2))) __bf16 v2bf;
__device__ __forceinline__ uint32_t pack2bf(float lo, float hi) {
  v2bf p = __builtin_amdgcn_cvt_pk_bf16_f32(lo, hi);   // v_cvt_pk_bf16_f32
  return __builtin_bit_cast(uint32_t, p);
}
#else
__device__ __forceinline__ uint32_t pack2bf(float lo, float hi) {
  return (uint32_t)f2bf(lo) | ((uint32_t)f2bf(hi) << 16);
}
#endif

static constexpr int Bn = 4;
static constexpr int Nn = 4096;   // 64*64
static constexpr int Cn = 256;
static constexpr int CqN = 32;

// Use the Tensor Data Mover for attention-tile staging when the builtin exists.
#if __has_builtin(__builtin_amdgcn_tensor_load_to_lds) && \
    __has_builtin(__builtin_amdgcn_s_wait_tensorcnt)
#define HAVE_TDM 1
#else
#define HAVE_TDM 0
#endif

// ---------------------------------------------------------------------------
// Kernel 1: q/k/v projections. One wave = one 16x16 output tile, K=C=256 in
// 8 bf16-WMMA steps. q,k stored row-major bf16 [B][N][32]; v stored transposed
// bf16 [B][C][N] so the AV GEMM's B-operand K-dim is contiguous.
// ---------------------------------------------------------------------------
__global__ __launch_bounds__(128) void proj_kernel(
    const float* __restrict__ x,
    const float* __restrict__ Wq, const float* __restrict__ bq,
    const float* __restrict__ Wk, const float* __restrict__ bk,
    const float* __restrict__ Wv, const float* __restrict__ bv,
    unsigned short* __restrict__ qws, unsigned short* __restrict__ kws,
    unsigned short* __restrict__ vtws)
{
  const int lane = threadIdx.x & 31;
  // job is wave-uniform: pin it to an SGPR so the mat/W/bias selection and the
  // epilogue store branch stay scalar (no exec-mask dance).
  const int job =
      __builtin_amdgcn_readfirstlane(blockIdx.x * 4 + (threadIdx.x >> 5));
  const int perB = 512 + 512 + 4096;             // q-tiles + k-tiles + v-tiles
  const int b  = job / perB;
  int r = job - b * perB;
  int mat, t;
  if (r < 512)        { mat = 0; t = r; }
  else if (r < 1024)  { mat = 1; t = r - 512; }
  else                { mat = 2; t = r - 1024; }
  int nt, dt, ldw;
  const float *W, *bias;
  if (mat == 2) { nt = t >> 4; dt = t & 15; W = Wv; bias = bv; ldw = Cn; }
  else          { nt = t >> 1; dt = t & 1;  ldw = CqN;
                  W = (mat == 0) ? Wq : Wk; bias = (mat == 0) ? bq : bk; }
  const int n0 = nt * 16, d0 = dt * 16;
  const int hl = lane >> 4;            // lane half: 0/1
  const int ll = lane & 15;
  const int rowA = n0 + ll;            // A-matrix row for this lane
  const int colB = d0 + ll;            // B/C/D column for this lane

  v8f acc = {};
  for (int kk = 0; kk < Cn; kk += 32) {
    // A tile 16x32: lane holds K = kk + hl*16 + {0..15}.
    const float* xrow = x + ((size_t)b * Nn + rowA) * Cn + kk + hl * 16;
    float4 f0 = *(const float4*)(xrow);
    float4 f1 = *(const float4*)(xrow + 4);
    float4 f2 = *(const float4*)(xrow + 8);
    float4 f3 = *(const float4*)(xrow + 12);
    V16BF a;
    a.w[0] = pack2bf(f0.x, f0.y); a.w[1] = pack2bf(f0.z, f0.w);
    a.w[2] = pack2bf(f1.x, f1.y); a.w[3] = pack2bf(f1.z, f1.w);
    a.w[4] = pack2bf(f2.x, f2.y); a.w[5] = pack2bf(f2.z, f2.w);
    a.w[6] = pack2bf(f3.x, f3.y); a.w[7] = pack2bf(f3.z, f3.w);
    // B tile 32x16: lane = column colB, K = kk + hl*16 + e (paired converts)
    const float* wp = W + (size_t)(kk + hl * 16) * ldw + colB;
    V16BF bm;
#pragma unroll
    for (int i = 0; i < 8; ++i)
      bm.w[i] = pack2bf(wp[(size_t)(2 * i) * ldw], wp[(size_t)(2 * i + 1) * ldw]);
    acc = __builtin_amdgcn_wmma_f32_16x16x32_bf16(false, a.v, false, bm.v,
                                                  (short)0, acc, false, false);
  }
  // Uniform epilogue: pick base + stride once, then 8 clean b16 stores.
  unsigned short* p0;
  size_t stride;
  if (mat == 2) {
    p0 = vtws + (size_t)b * Cn * Nn + (size_t)colB * Nn + (n0 + hl * 8);
    stride = 1;
  } else {
    unsigned short* o = (mat == 0) ? qws : kws;
    p0 = o + ((size_t)b * Nn + (n0 + hl * 8)) * CqN + colB;
    stride = CqN;
  }
  const float bb = bias[colB];
#pragma unroll
  for (int rr = 0; rr < 8; ++rr) p0[(size_t)rr * stride] = f2bf(acc[rr] + bb);
}

// ---------------------------------------------------------------------------
// Kernel 2: energy = q . k^T  (K = Cq = 32 -> exactly one WMMA per tile).
// One wave = one 16-row i-tile across a 256-column j-group (A reused 16x).
// Raw scores written straight into the attention output region.
// ---------------------------------------------------------------------------
__global__ __launch_bounds__(128) void energy_kernel(
    const unsigned short* __restrict__ qws,
    const unsigned short* __restrict__ kws,
    float* __restrict__ attn)
{
  const int lane = threadIdx.x & 31;
  const int job =
      __builtin_amdgcn_readfirstlane(blockIdx.x * 4 + (threadIdx.x >> 5));
  const int b  = job >> 12;
  const int rm = job & 4095;
  const int i0 = (rm >> 4) * 16;
  const int jg = rm & 15;
  const int hl = lane >> 4, ll = lane & 15;

  // A = q rows i0..i0+15 in WMMA A-layout (pairs are contiguous in Cq)
  V16BF a;
  const unsigned short* qrow = qws + ((size_t)b * Nn + (i0 + ll)) * CqN;
#pragma unroll
  for (int i = 0; i < 8; ++i) {
    const int pb = 2 * i + hl * 16;
    a.w[i] = *(const uint32_t*)(qrow + pb);
  }
  float* out = attn + (size_t)b * Nn * Nn;
  for (int jt = 0; jt < 16; ++jt) {
    const int j0 = jg * 256 + jt * 16;
    const unsigned short* krow =
        kws + ((size_t)b * Nn + (j0 + ll)) * CqN + hl * 16;
    V16BF bm;
#pragma unroll
    for (int i = 0; i < 8; ++i) bm.w[i] = *(const uint32_t*)(krow + 2 * i);
    v8f c = {};
    c = __builtin_amdgcn_wmma_f32_16x16x32_bf16(false, a.v, false, bm.v,
                                                (short)0, c, false, false);
#pragma unroll
    for (int rr = 0; rr < 8; ++rr) {
      const int row = i0 + rr + hl * 8;
      out[(size_t)row * Nn + (j0 + ll)] = c[rr];
    }
  }
}

// ---------------------------------------------------------------------------
// Kernel 3: in-place row softmax over 4096 keys. One block per row; row held
// in registers (16 f32/thread), LDS tree reductions for max and sum.
// ---------------------------------------------------------------------------
__global__ __launch_bounds__(256) void softmax_kernel(float* __restrict__ attn)
{
  __shared__ float red[256];
  const int t = threadIdx.x;
  float* row = attn + (size_t)blockIdx.x * Nn;
  float4 v[4];
#pragma unroll
  for (int i = 0; i < 4; ++i) v[i] = *(const float4*)(row + (i * 256 + t) * 4);
  float m = -3.4e38f;
#pragma unroll
  for (int i = 0; i < 4; ++i)
    m = fmaxf(m, fmaxf(fmaxf(v[i].x, v[i].y), fmaxf(v[i].z, v[i].w)));
  red[t] = m; __syncthreads();
  for (int s = 128; s > 0; s >>= 1) {
    if (t < s) red[t] = fmaxf(red[t], red[t + s]);
    __syncthreads();
  }
  m = red[0]; __syncthreads();
  float sum = 0.f;
#pragma unroll
  for (int i = 0; i < 4; ++i) {
    v[i].x = __expf(v[i].x - m); v[i].y = __expf(v[i].y - m);
    v[i].z = __expf(v[i].z - m); v[i].w = __expf(v[i].w - m);
    sum += v[i].x + v[i].y + v[i].z + v[i].w;
  }
  red[t] = sum; __syncthreads();
  for (int s = 128; s > 0; s >>= 1) {
    if (t < s) red[t] += red[t + s];
    __syncthreads();
  }
  const float inv = 1.0f / red[0];
#pragma unroll
  for (int i = 0; i < 4; ++i) {
    v[i].x *= inv; v[i].y *= inv; v[i].z *= inv; v[i].w *= inv;
    *(float4*)(row + (i * 256 + t) * 4) = v[i];
  }
}

// ---------------------------------------------------------------------------
// Kernel 4: out = attn @ v, y = gamma*out + x. Workgroup (8 waves) owns one
// 16-row i-strip x all 256 channels. The 16x32 attention chunk is staged into
// LDS once per K-step — via the Tensor Data Mover (2D tile descriptor, one
// issuing wave, s_wait_tensorcnt fence) when available — and shared by all 8
// waves; each wave does two 16-col tiles -> 2 WMMA per step (256 WMMA/wave).
// ---------------------------------------------------------------------------
__global__ __launch_bounds__(256) void av_kernel(
    const float* __restrict__ attn,
    const unsigned short* __restrict__ vtws,
    const float* __restrict__ xin,
    const float* __restrict__ gammap,
    float* __restrict__ y)
{
#if HAVE_TDM
  __shared__ float lds_f[16 * 32];        // 16x32 f32 tile, row-major (2KB)
#else
  __shared__ uint32_t lds_a[256];         // 16x32 bf16 tile, row-major pairs
#endif
  const int tid  = threadIdx.x;
  const int lane = tid & 31;
  const int wave = __builtin_amdgcn_readfirstlane(tid >> 5);  // scalar wave id
  const int b  = blockIdx.x >> 8;
  const int i0 = (blockIdx.x & 255) * 16;
  const int hl = lane >> 4, ll = lane & 15;
  const int c0 = wave * 32;               // this wave's two column tiles
  const float* arow_base = attn + ((size_t)b * Nn + i0) * Nn;
#if !HAVE_TDM
  const int srr = (tid * 2) >> 5;         // staging row for this thread
  const int scc = (tid * 2) & 31;         // staging col pair
#endif

  v8f acc0 = {}, acc1 = {};
  for (int j0 = 0; j0 < Nn; j0 += 32) {
#if HAVE_TDM
    if (wave == 0) {
      // Tensor DMA descriptor (ISA 8.3-8.5): 2D tile 32x16 f32 out of a
      // row-major [Nn,Nn] tensor into LDS.
      const uint64_t ga = (uint64_t)(uintptr_t)(arow_base + j0);
      const uint32_t lb = (uint32_t)(uintptr_t)(&lds_f[0]);
      u32x4 g0;
      g0.x = 1u;                                        // count=1 (valid)
      g0.y = lb;                                        // lds_addr (bytes)
      g0.z = (uint32_t)ga;                              // global_addr[31:0]
      g0.w = (uint32_t)((ga >> 32) & 0x01FFFFFFu)       // global_addr[56:32]
             | 0x80000000u;                             // type=2 ("image")
      i32x8 g1;
      g1[0] = (int)(2u << 16);          // data_size=2 (4 bytes)
      g1[1] = (int)((uint32_t)Nn << 16);// tensor_dim0[15:0] in [31:16]
      g1[2] = (int)((uint32_t)Nn << 16);// dim0 hi=0 | tensor_dim1[15:0]<<16
      g1[3] = (int)(32u << 16);         // dim1 hi=0 | tile_dim0=32
      g1[4] = 16;                       // tile_dim1=16, tile_dim2=0
      g1[5] = Nn;                       // tensor_dim0_stride[31:0]
      g1[6] = 0;                        // stride hi / dim1_stride lo
      g1[7] = 0;
      const i32x4 gz = {0, 0, 0, 0};
#if __clang_major__ >= 23
      const i32x8 gz8 = {0, 0, 0, 0, 0, 0, 0, 0};
      __builtin_amdgcn_tensor_load_to_lds(g0, g1, gz, gz, gz8, 0);
#else
      __builtin_amdgcn_tensor_load_to_lds(g0, g1, gz, gz, 0);
#endif
      __builtin_amdgcn_s_wait_tensorcnt(0);
    }
    __syncthreads();
    // A operand from the f32 LDS tile: per-lane ds_load_b64 pairs + packed cvt.
    V16BF a;
#pragma unroll
    for (int i = 0; i < 8; ++i) {
      const int pb = 2 * i + hl * 16;
      float2 f = *(const float2*)(&lds_f[ll * 32 + pb]);
      a.w[i] = pack2bf(f.x, f.y);
    }
#else
    // Fallback staging: threads convert f32->bf16 into LDS cooperatively.
    {
      float2 f = *(const float2*)(arow_base + (size_t)srr * Nn + j0 + scc);
      lds_a[tid] = pack2bf(f.x, f.y);
      if (j0 + 32 < Nn)
        __builtin_prefetch(arow_base + (size_t)srr * Nn + j0 + 32 + scc, 0, 0);
    }
    __syncthreads();
    V16BF a;
#pragma unroll
    for (int i = 0; i < 8; ++i) {
      const int pi = i + hl * 8;  // uint-pair index
      a.w[i] = lds_a[ll * 16 + pi];
    }
#endif
    // B tiles from transposed v: K(=j) contiguous per lane-column
    const unsigned short* vb0 =
        vtws + (size_t)b * Cn * Nn + (size_t)(c0 + ll) * Nn + j0 + hl * 16;
    const unsigned short* vb1 = vb0 + (size_t)16 * Nn;
    V16BF b0, b1;
#pragma unroll
    for (int i = 0; i < 8; ++i) {
      b0.w[i] = *(const uint32_t*)(vb0 + 2 * i);
      b1.w[i] = *(const uint32_t*)(vb1 + 2 * i);
    }
    acc0 = __builtin_amdgcn_wmma_f32_16x16x32_bf16(false, a.v, false, b0.v,
                                                   (short)0, acc0, false, false);
    acc1 = __builtin_amdgcn_wmma_f32_16x16x32_bf16(false, a.v, false, b1.v,
                                                   (short)0, acc1, false, false);
    __syncthreads();
  }
  const float g = *gammap;
#pragma unroll
  for (int rr = 0; rr < 8; ++rr) {
    const int row = i0 + rr + hl * 8;
    const size_t base = ((size_t)b * Nn + row) * Cn;
    const int ca = c0 + ll, cb = ca + 16;
    y[base + ca] = acc0[rr] * g + xin[base + ca];
    y[base + cb] = acc1[rr] * g + xin[base + cb];
  }
}

// ---------------------------------------------------------------------------
extern "C" void kernel_launch(void* const* d_in, const int* in_sizes, int n_in,
                              void* d_out, int out_size, void* d_ws, size_t ws_size,
                              hipStream_t stream) {
  (void)in_sizes; (void)n_in; (void)out_size; (void)ws_size;
  const float* x     = (const float*)d_in[0];
  const float* Wq    = (const float*)d_in[1];
  const float* bq    = (const float*)d_in[2];
  const float* Wk    = (const float*)d_in[3];
  const float* bk    = (const float*)d_in[4];
  const float* Wv    = (const float*)d_in[5];
  const float* bv    = (const float*)d_in[6];
  const float* gamma = (const float*)d_in[7];

  float* y    = (float*)d_out;                       // [B,N,C]
  float* attn = y + (size_t)Bn * Nn * Cn;            // [B,N,N]

  unsigned short* qws  = (unsigned short*)d_ws;              // [B,N,32] bf16
  unsigned short* kws  = qws + (size_t)Bn * Nn * CqN;        // [B,N,32] bf16
  unsigned short* vtws = kws + (size_t)Bn * Nn * CqN;        // [B,C,N] bf16

  // 20480 wave-tiles / 4 waves per block
  proj_kernel<<<dim3(5120), dim3(128), 0, stream>>>(x, Wq, bq, Wk, bk, Wv, bv,
                                                    qws, kws, vtws);
  // 16384 wave-jobs / 4 waves per block
  energy_kernel<<<dim3(4096), dim3(128), 0, stream>>>(qws, kws, attn);
  // one block per attention row
  softmax_kernel<<<dim3(16384), dim3(256), 0, stream>>>(attn);
  // one block per (b, i-strip)
  av_kernel<<<dim3(1024), dim3(256), 0, stream>>>(attn, vtws, x, gamma, y);
}
